// MLPAttentionWithQuery_28604482191941
// MI455X (gfx1250) — compile-verified, matrix operand order checked by reading, not run on hardware
//
#include <hip/hip_runtime.h>

// ---------------------------------------------------------------------------
// MLP attention with query, CDNA5 (gfx1250, wave32, WMMA bf16 path)
//   B=64, L=2048, E=512, Q=256
//   h = [enc | q] ; u = tanh(h@W1 + b1) ; logits = u.v ; masked softmax ;
//   context = attn @ enc
// Split: h@W1 = enc@W1e + q@W1q ; q@W1q + b1 folded into per-batch bias qb.
// ---------------------------------------------------------------------------

#define B_DIM 64
#define L_DIM 2048
#define E_DIM 512
#define Q_DIM 256
#define NEG_VAL (-1.0e9f)
#define EPS_VAL (1.0e-5f)

typedef __attribute__((ext_vector_type(16))) __bf16 v16bf;
typedef __attribute__((ext_vector_type(2)))  __bf16 v2bf;
typedef __attribute__((ext_vector_type(8)))  float  v8f;

#define WMMA_BF16(A, Bf, C) \
    __builtin_amdgcn_wmma_f32_16x16x32_bf16(false, (A), false, (Bf), (short)0, (C), false, false)

// ---------------------------------------------------------------------------
// Kernel 0: convert W1e (rows 0..511 of W1, [768,512] row-major) to bf16,
// pre-swizzled into per-lane WMMA B fragments:
//   frag index f = (ntile*16 + ks)*32 + lane, 16 bf16 elements per lane.
//   lane: N = ntile*16 + (lane&15); kbase = ks*32 + (lane>=16 ? 8 : 0)
//   elem j: K = kbase + j + (j>=8 ? 8 : 0)   (ISA 16-bit matrix packing)
// ---------------------------------------------------------------------------
__global__ __launch_bounds__(256) void k_swz(const float* __restrict__ W1,
                                             __bf16* __restrict__ w1sw)
{
    const int id   = blockIdx.x * 256 + threadIdx.x;   // 0 .. 262143
    const int j    = id & 15;
    const int lane = (id >> 4) & 31;
    const int ks   = (id >> 9) & 15;
    const int nt   = id >> 13;                          // 0..31
    const int N    = nt * 16 + (lane & 15);
    const int kb   = ks * 32 + ((lane >> 4) & 1) * 8;
    const int K    = kb + j + ((j >= 8) ? 8 : 0);
    w1sw[id] = (__bf16)W1[(size_t)K * E_DIM + N];
}

// ---------------------------------------------------------------------------
// Kernel 1: qb[b,e] = b1[e] + sum_c query[b,c] * W1[512+c, e]
// ---------------------------------------------------------------------------
__global__ __launch_bounds__(256) void k_qproj(const float* __restrict__ query,
                                               const float* __restrict__ W1,
                                               const float* __restrict__ b1,
                                               float* __restrict__ qb)
{
    __shared__ float q[Q_DIM];
    const int b = blockIdx.x, t = threadIdx.x;
    q[t] = query[(size_t)b * Q_DIM + t];
    __syncthreads();
    float a0 = b1[t], a1 = b1[t + 256];
    for (int c = 0; c < Q_DIM; ++c) {
        const float qc = q[c];
        const float* wrow = W1 + (size_t)(E_DIM + c) * E_DIM;
        a0 += qc * wrow[t];
        a1 += qc * wrow[t + 256];
    }
    qb[(size_t)b * E_DIM + t]       = a0;
    qb[(size_t)b * E_DIM + t + 256] = a1;
}

// ---------------------------------------------------------------------------
// Kernel 2: main GEMM + tanh + logits.
//   block: (b, 32-row L tile). A tile staged ONCE in LDS as bf16 in WMMA
//   A-fragment layout (removes the 8x redundant global read + cvt).
//   8 waves; wave w owns N-tiles w*4..w*4+3 and both 16-row M-tiles:
//   per k-step 2 LDS A-frag reads + 4 global B-frag reads -> 8 WMMAs.
//   K loop unroll capped at 2: ~180 live VGPRs (64 acc + 2x48 fragments),
//   stays under the 256 direct-address limit -> no scratch spills, decent
//   occupancy; deeper latency hiding comes from 8 waves/block.
// ---------------------------------------------------------------------------
__global__ __launch_bounds__(256) void k_gemm(const float* __restrict__ enc,
                                              const __bf16* __restrict__ w1sw,
                                              const float* __restrict__ qb,
                                              const float* __restrict__ vvec,
                                              float* __restrict__ logits)
{
    __shared__ __bf16 sA[2 * 16 * 32 * 16];   // 2 M-tiles x 16 ks x 32 lanes x 16 = 32 KB
    __shared__ float  red[8][32];

    const int tid  = threadIdx.x;
    const int wave = tid >> 5;
    const int lane = tid & 31;
    const int half = lane >> 4;       // selects K+0 / K+8 sub-block
    const int lrow = lane & 15;       // M row (A) / N col (B,C,D)
    const int mt = blockIdx.x % (L_DIM / 32);
    const int b  = blockIdx.x / (L_DIM / 32);
    const int l0 = mt * 32;

    // ---- Stage A tile (32 rows x 512 K, f32) into LDS as bf16 fragments.
    // Forward frag mapping: row = mt2*16 + (lane&15); K = ks*32 + half*8 + j + (j>=8?8:0)
    // Inverse (from k): ks=k>>5; krem=k&31; half=(krem>>3)&1; j=(krem&7)+(krem>=16?8:0)
    {
        const float* atile = enc + ((size_t)b * L_DIM + l0) * E_DIM;
#pragma unroll
        for (int i = 0; i < 32; ++i) {
            const int flat = (i * 256 + tid) * 2;      // even flat elem, 0..16382
            const int r    = flat >> 9;                // row 0..31
            const int k    = flat & 511;
            const int ks   = k >> 5;
            const int krem = k & 31;
            const int hh   = (krem >> 3) & 1;
            const int j    = (krem & 7) + ((krem >= 16) ? 8 : 0);
            const int ln   = hh * 16 + (r & 15);
            const int mt2  = r >> 4;
            const float* ap = atile + (size_t)r * E_DIM + k;
            if ((i & 3) == 0) __builtin_prefetch(ap + 2048, 0, 1);  // ~4 iter ahead
            const float2 f = *(const float2*)ap;
            v2bf p; p[0] = (__bf16)f.x; p[1] = (__bf16)f.y;
            *(v2bf*)(&sA[((mt2 * 16 + ks) * 32 + ln) * 16 + j]) = p;
        }
    }
    __syncthreads();

    // ---- Main K loop: all fragment addresses are base + immediate offset.
    const v16bf* sAf = (const v16bf*)sA;               // one v16bf per lane-frag
    const char*  bb  = (const char*)w1sw + ((size_t)(wave * 4) * 512 + lane) * 32;

    v8f c00 = {}, c01 = {}, c02 = {}, c03 = {};
    v8f c10 = {}, c11 = {}, c12 = {}, c13 = {};

#pragma unroll 2
    for (int ks = 0; ks < 16; ++ks) {
        const v16bf b0 = *(const v16bf*)(bb);
        const v16bf b1 = *(const v16bf*)(bb + 16384);  // +16 K frags (1 N-tile)
        const v16bf b2 = *(const v16bf*)(bb + 32768);
        const v16bf b3 = *(const v16bf*)(bb + 49152);
        const v16bf a0 = sAf[ks * 32 + lane];          // M-tile 0
        const v16bf a1 = sAf[(16 + ks) * 32 + lane];   // M-tile 1
        bb += 1024;                                    // next ks (32 frags)

        c00 = WMMA_BF16(a0, b0, c00);
        c01 = WMMA_BF16(a0, b1, c01);
        c02 = WMMA_BF16(a0, b2, c02);
        c03 = WMMA_BF16(a0, b3, c03);
        c10 = WMMA_BF16(a1, b0, c10);
        c11 = WMMA_BF16(a1, b1, c11);
        c12 = WMMA_BF16(a1, b2, c12);
        c13 = WMMA_BF16(a1, b3, c13);
    }

    // ---- Epilogue: C layout -> lane elem r is row M = r + half*8 (+16 for
    // M-tile 1), col N = ntile*16 + lrow. Bias + tanh + * v[N], per-row sums.
    float rp0[8], rp1[8];
#pragma unroll
    for (int r = 0; r < 8; ++r) { rp0[r] = 0.0f; rp1[r] = 0.0f; }

#define EPILOG(C0, C1, NT)                                                    \
    {                                                                         \
        const int   ncol = (wave * 4 + (NT)) * 16 + lrow;                     \
        const float qv   = qb[(size_t)b * E_DIM + ncol];                      \
        const float vv   = vvec[ncol];                                        \
        _Pragma("unroll")                                                     \
        for (int r = 0; r < 8; ++r) {                                         \
            rp0[r] += tanhf(C0[r] + qv) * vv;                                 \
            rp1[r] += tanhf(C1[r] + qv) * vv;                                 \
        }                                                                     \
    }
    EPILOG(c00, c10, 0)
    EPILOG(c01, c11, 1)
    EPILOG(c02, c12, 2)
    EPILOG(c03, c13, 3)
#undef EPILOG

    // Deterministic reduction across the 16 lanes of each half (xor masks
    // 1,2,4,8 never cross bit 4), then across 8 waves via LDS.
#pragma unroll
    for (int r = 0; r < 8; ++r) {
        float s0 = rp0[r], s1 = rp1[r];
        s0 += __shfl_xor(s0, 1, 32); s1 += __shfl_xor(s1, 1, 32);
        s0 += __shfl_xor(s0, 2, 32); s1 += __shfl_xor(s1, 2, 32);
        s0 += __shfl_xor(s0, 4, 32); s1 += __shfl_xor(s1, 4, 32);
        s0 += __shfl_xor(s0, 8, 32); s1 += __shfl_xor(s1, 8, 32);
        rp0[r] = s0; rp1[r] = s1;
    }
    if (lrow == 0) {                      // lanes 0 and 16
#pragma unroll
        for (int r = 0; r < 8; ++r) {
            red[wave][half * 8 + r]      = rp0[r];
            red[wave][16 + half * 8 + r] = rp1[r];
        }
    }
    __syncthreads();
    if (tid < 32) {
        float s = 0.0f;
#pragma unroll
        for (int w = 0; w < 8; ++w) s += red[w][tid];
        logits[(size_t)b * L_DIM + l0 + tid] = s;
    }
}

// ---------------------------------------------------------------------------
// Kernel 3: per-batch masked softmax over L + context = attn @ enc.
// ---------------------------------------------------------------------------
__global__ __launch_bounds__(256) void k_softmax_ctx(const float* __restrict__ enc,
                                                     const int* __restrict__ length,
                                                     const float* __restrict__ logits,
                                                     float* __restrict__ ctx,
                                                     float* __restrict__ attn)
{
    __shared__ float sa[L_DIM];
    __shared__ float sr[256];
    const int b = blockIdx.x, t = threadIdx.x;
    const int len = length[b];
    const float* lg = logits + (size_t)b * L_DIM;

    float m = -3.4e38f;
    for (int l = t; l < L_DIM; l += 256) {
        const float x = (l < len) ? lg[l] : NEG_VAL;
        sa[l] = x;
        m = fmaxf(m, x);
    }
    sr[t] = m; __syncthreads();
    for (int s = 128; s > 0; s >>= 1) { if (t < s) sr[t] = fmaxf(sr[t], sr[t + s]); __syncthreads(); }
    m = sr[0]; __syncthreads();

    float sum = 0.0f;
    for (int l = t; l < L_DIM; l += 256) {
        const float e = (l < len) ? __expf(sa[l] - m) : 0.0f;
        sa[l] = e;
        sum += e;
    }
    sr[t] = sum; __syncthreads();
    for (int s = 128; s > 0; s >>= 1) { if (t < s) sr[t] += sr[t + s]; __syncthreads(); }
    const float inv = 1.0f / (sr[0] + EPS_VAL);
    __syncthreads();

    for (int l = t; l < L_DIM; l += 256) {
        const float a = sa[l] * inv;
        sa[l] = a;
        attn[(size_t)b * L_DIM + l] = a;
    }
    __syncthreads();

    // context: thread t owns columns e = 2t, 2t+1 (coalesced float2 rows)
    const float2* eb = (const float2*)(enc + (size_t)b * L_DIM * E_DIM);
    float a0 = 0.0f, a1 = 0.0f;
    for (int l = 0; l < L_DIM; ++l) {
        const float a = sa[l];
        const float2 ev = eb[(size_t)l * (E_DIM / 2) + t];
        a0 += a * ev.x;
        a1 += a * ev.y;
    }
    ctx[(size_t)b * E_DIM + 2 * t]     = a0;
    ctx[(size_t)b * E_DIM + 2 * t + 1] = a1;
}

// ---------------------------------------------------------------------------
extern "C" void kernel_launch(void* const* d_in, const int* in_sizes, int n_in,
                              void* d_out, int out_size, void* d_ws, size_t ws_size,
                              hipStream_t stream)
{
    const float* enc   = (const float*)d_in[0];   // [64,2048,512]
    const float* query = (const float*)d_in[1];   // [64,256]
    const int*   len   = (const int*)  d_in[2];   // [64]
    const float* W1    = (const float*)d_in[3];   // [768,512]
    const float* b1    = (const float*)d_in[4];   // [512]
    const float* v     = (const float*)d_in[5];   // [512]

    float* out  = (float*)d_out;
    float* ctx  = out;                       // [64,512]
    float* attn = out + B_DIM * E_DIM;       // [64,2048]

    char* ws = (char*)d_ws;
    __bf16* w1sw   = (__bf16*)ws;                                   // 512 KB
    float*  qb     = (float*)(ws + (size_t)E_DIM * E_DIM * 2);      // 128 KB
    float*  logits = (float*)(ws + (size_t)E_DIM * E_DIM * 2
                                 + (size_t)B_DIM * E_DIM * 4);      // 512 KB

    k_swz<<<(E_DIM * E_DIM) / 256, 256, 0, stream>>>(W1, w1sw);
    k_qproj<<<B_DIM, 256, 0, stream>>>(query, W1, b1, qb);
    k_gemm<<<B_DIM * (L_DIM / 32), 256, 0, stream>>>(enc, w1sw, qb, v, logits);
    k_softmax_ctx<<<B_DIM, 256, 0, stream>>>(enc, len, logits, ctx, attn);
}
